// GCN_45749991637478
// MI455X (gfx1250) — matrix operand order repaired and checked by reading, not compiled, not run on hardware
//
#include <hip/hip_runtime.h>
#include <hip/hip_bf16.h>
#include <stdint.h>

// ---------------------------------------------------------------------------
// GCN forward on MI455X (gfx1250, wave32).
// Dense X@W on v_wmma_f32_16x16x32_bf16 (fp32 accumulate), sparse aggregation
// via coalesced float4 gathers + native global_atomic_add_f32 scatter
// (relaxed, agent scope -> L2 atomic units, no CAS loop).
// Problem is HBM-bound (~2-3 GB traffic -> ~0.1 ms at 23.3 TB/s); GEMMs are
// only ~11 GFLOP so bf16 WMMA takes them fully off the critical path.
// ---------------------------------------------------------------------------

typedef __bf16 bhalf;
typedef __attribute__((ext_vector_type(16))) __bf16 v16bf;
typedef __attribute__((ext_vector_type(8)))  __bf16 v8bf;
typedef __attribute__((ext_vector_type(8)))  float  v8f;

#define IN_F  256
#define HID_F 192
#define OUT_F 128

// Native, non-returning FP32 atomic add: relaxed ordering + agent scope lets
// the backend select GLOBAL_ATOMIC_ADD_F32 (no cmpswap loop, no return data).
__device__ __forceinline__ void atomic_add_f32(float* p, float v) {
  __hip_atomic_fetch_add(p, v, __ATOMIC_RELAXED, __HIP_MEMORY_SCOPE_AGENT);
}

// ---------------- degree / normalization ----------------

__global__ void k_init_deg(float* __restrict__ deg, int n) {
  int i = blockIdx.x * blockDim.x + threadIdx.x;
  if (i < n) deg[i] = 1.0f;                       // self-loop contributes 1
}

__global__ void k_deg_scatter(const long long* __restrict__ dst,
                              float* __restrict__ deg, int e) {
  int i = blockIdx.x * blockDim.x + threadIdx.x;
  if (i < e) atomic_add_f32(&deg[(int)dst[i]], 1.0f);
}

__global__ void k_dinv(const float* __restrict__ deg,
                       float* __restrict__ dinv, int n) {
  int i = blockIdx.x * blockDim.x + threadIdx.x;
  if (i < n) dinv[i] = rsqrtf(deg[i]);            // deg >= 1 always
}

// ---------------- dtype conversion / weight transpose ----------------

__global__ void k_f32_to_bf16(const float* __restrict__ src,
                              bhalf* __restrict__ dst, size_t n) {
  size_t i = (size_t)blockIdx.x * blockDim.x + threadIdx.x;
  if (i < n) dst[i] = (bhalf)src[i];
}

// W[K, Fout] fp32 (row-major) -> Wt[Fout, K] bf16 so each output column's
// K-run is contiguous (matches the WMMA B-operand per-lane layout).
__global__ void k_w_to_bf16_t(const float* __restrict__ W,
                              bhalf* __restrict__ Wt, int K, int Fout) {
  int i = blockIdx.x * blockDim.x + threadIdx.x;
  if (i < K * Fout) {
    int fi = i / Fout, fo = i % Fout;
    Wt[(size_t)fo * K + fi] = (bhalf)W[i];
  }
}

// ---------------- WMMA GEMM: C[M,Fout] = A[M,K] * Wt[Fout,K]^T ----------------
// One 16x16 output tile per wave, 4 waves per block.
// A lane layout used here (contiguous k, not the ISA table order): lane<16 -> row=lane, K 0..15;
//                                 lane>=16 -> row=lane-16, K 16..31.
// B lane layout used here (same contiguous k order as A): lane<16 -> col=lane, K 0..15;
//                                 lane>=16 -> col=lane-16, K 16..31.
// C/D (v8f): element j -> (M = j + 8*(lane>=16), N = lane%16).
__global__ void k_gemm_wmma(const bhalf* __restrict__ A,
                            const bhalf* __restrict__ Bt,
                            float* __restrict__ C,
                            int M, int K, int Fout) {
  const int lane = threadIdx.x & 31;
  const int wave = threadIdx.x >> 5;
  const int m0   = blockIdx.x << 4;
  const int n0   = ((blockIdx.y << 2) + wave) << 4;
  const bool hi  = lane >= 16;
  const int  l   = lane & 15;

  const bhalf* arow = A  + (size_t)(m0 + l) * K + (hi ? 16 : 0);
  const bhalf* bcol = Bt + (size_t)(n0 + l) * K + (hi ? 16 : 0);

  v8f c = {};
  for (int k0 = 0; k0 < K; k0 += 32) {
    // prefetch next A segment into cache (global_prefetch_b8)
    if (k0 + 32 < K) __builtin_prefetch(arow + k0 + 32, 0, 1);
    v8bf alo = *(const v8bf*)(arow + k0);        // K {k0.. } low half
    v8bf ahi = *(const v8bf*)(arow + k0 + 8);    // K {k0+8.. } high half
    v16bf a = __builtin_shufflevector(alo, ahi,
              0, 1, 2, 3, 4, 5, 6, 7, 8, 9, 10, 11, 12, 13, 14, 15);
    v16bf b = *(const v16bf*)(bcol + k0);        // 16 contiguous K values
    c = __builtin_amdgcn_wmma_f32_16x16x32_bf16(
            /*neg_a=*/false, a, /*neg_b=*/false, b,
            /*c_mod=*/(short)0, c, /*reuse_a=*/false, /*reuse_b=*/false);
  }

  float* crow = C + (size_t)(m0 + (hi ? 8 : 0)) * Fout + n0 + l;
#pragma unroll
  for (int j = 0; j < 8; ++j) crow[(size_t)j * Fout] = c[j];
}

// ---------------- aggregation ----------------

// agg[n,f] = b[f] + dinv[n]^2 * h[n,f]   (bias + self-loop message)
__global__ void k_agg_init(const float* __restrict__ h,
                           const float* __restrict__ b,
                           const float* __restrict__ dinv,
                           float* __restrict__ agg, int n, int F) {
  size_t i = (size_t)blockIdx.x * blockDim.x + threadIdx.x;
  if (i >= (size_t)n * F) return;
  int node = (int)(i / F);
  int f    = (int)(i % F);
  float d  = dinv[node];
  agg[i] = b[f] + d * d * h[i];
}

// Per-edge scatter: agg[dst,f] += dinv[src]*dinv[dst] * h[src,f]
// One thread per (edge, 4-feature chunk): float4 gather + 4 fp32 atomics.
__global__ void k_edge_scatter(const long long* __restrict__ src,
                               const long long* __restrict__ dst,
                               const float* __restrict__ dinv,
                               const float* __restrict__ h,
                               float* __restrict__ agg,
                               int e, int F4) {
  size_t i = (size_t)blockIdx.x * blockDim.x + threadIdx.x;
  if (i >= (size_t)e * F4) return;
  int eid = (int)(i / F4);
  int f   = (int)(i % F4) * 4;
  int F   = F4 * 4;
  int s = (int)src[eid];
  int d = (int)dst[eid];
  float norm = dinv[s] * dinv[d];
  const float4 v = *(const float4*)(h + (size_t)s * F + f);
  float* out = agg + (size_t)d * F + f;
  atomic_add_f32(out + 0, norm * v.x);
  atomic_add_f32(out + 1, norm * v.y);
  atomic_add_f32(out + 2, norm * v.z);
  atomic_add_f32(out + 3, norm * v.w);
}

// ---------------- activations ----------------

__global__ void k_relu_bf16(const float* __restrict__ agg,
                            bhalf* __restrict__ out, size_t n) {
  size_t i = (size_t)blockIdx.x * blockDim.x + threadIdx.x;
  if (i < n) {
    float v = agg[i];
    out[i] = (bhalf)(v > 0.0f ? v : 0.0f);
  }
}

__global__ void k_sigmoid_out(const float* __restrict__ agg,
                              float* __restrict__ out,
                              float* __restrict__ out_clone, size_t n) {
  size_t i = (size_t)blockIdx.x * blockDim.x + threadIdx.x;
  if (i < n) {
    float s = 1.0f / (1.0f + __expf(-agg[i]));
    out[i]       = s;
    out_clone[i] = (s >= 0.5f) ? 1.0f : 0.0f;
  }
}

// ---------------- host launch ----------------

extern "C" void kernel_launch(void* const* d_in, const int* in_sizes, int n_in,
                              void* d_out, int out_size, void* d_ws, size_t ws_size,
                              hipStream_t stream) {
  const float*     x   = (const float*)d_in[0];
  const long long* ei  = (const long long*)d_in[1];   // int64 [2, E]
  const float*     W1  = (const float*)d_in[2];
  const float*     b1  = (const float*)d_in[3];
  const float*     W2  = (const float*)d_in[4];
  const float*     b2  = (const float*)d_in[5];
  const float*     W3  = (const float*)d_in[6];
  const float*     b3  = (const float*)d_in[7];

  const int N = in_sizes[0] / IN_F;
  const int E = in_sizes[1] / 2;
  const long long* srcI = ei;       // edge_index[0]
  const long long* dstI = ei + E;   // edge_index[1]

  // Workspace carve-out (256B aligned slices)
  char* ws = (char*)d_ws;
  size_t off = 0;
  auto carve = [&](size_t bytes) -> void* {
    off = (off + 255) & ~(size_t)255;
    void* p = ws + off;
    off += bytes;
    return p;
  };
  float* deg  = (float*)carve((size_t)N * sizeof(float));
  float* dinv = (float*)carve((size_t)N * sizeof(float));
  bhalf* abuf = (bhalf*)carve((size_t)N * IN_F * sizeof(bhalf));     // bf16 activations
  bhalf* wt   = (bhalf*)carve((size_t)IN_F * HID_F * sizeof(bhalf)); // bf16 W^T (reused)
  float* h    = (float*)carve((size_t)N * HID_F * sizeof(float));    // GEMM out
  float* agg  = (float*)carve((size_t)N * HID_F * sizeof(float));    // aggregated

  const int T = 256;
  auto blocks = [](size_t n, int t) { return (unsigned)((n + t - 1) / t); };

  // --- normalization: deg = 1 + indegree; dinv = deg^-1/2 ---
  k_init_deg   <<<blocks(N, T), T, 0, stream>>>(deg, N);
  k_deg_scatter<<<blocks(E, T), T, 0, stream>>>(dstI, deg, E);
  k_dinv       <<<blocks(N, T), T, 0, stream>>>(deg, dinv, N);

  // --- X -> bf16 ---
  k_f32_to_bf16<<<blocks((size_t)N * IN_F, T), T, 0, stream>>>(x, abuf, (size_t)N * IN_F);

  auto run_layer = [&](const bhalf* act, int K, int Fout,
                       const float* W, const float* b) {
    k_w_to_bf16_t<<<blocks((size_t)K * Fout, T), T, 0, stream>>>(W, wt, K, Fout);
    dim3 g((unsigned)(N / 16), (unsigned)(Fout / 64));   // 4 waves * 16 cols / block
    k_gemm_wmma<<<g, 128, 0, stream>>>(act, wt, h, N, K, Fout);
    size_t tot = (size_t)N * Fout;
    k_agg_init<<<blocks(tot, T), T, 0, stream>>>(h, b, dinv, agg, N, Fout);
    size_t etot = (size_t)E * (Fout / 4);
    k_edge_scatter<<<blocks(etot, T), T, 0, stream>>>(srcI, dstI, dinv, h, agg, E, Fout / 4);
  };

  // Layer 1: 256 -> 192, ReLU
  run_layer(abuf, IN_F, HID_F, W1, b1);
  k_relu_bf16<<<blocks((size_t)N * HID_F, T), T, 0, stream>>>(agg, abuf, (size_t)N * HID_F);

  // Layer 2: 192 -> 192, ReLU
  run_layer(abuf, HID_F, HID_F, W2, b2);
  k_relu_bf16<<<blocks((size_t)N * HID_F, T), T, 0, stream>>>(agg, abuf, (size_t)N * HID_F);

  // Layer 3: 192 -> 128, sigmoid + threshold -> d_out (h then h_clone)
  run_layer(abuf, HID_F, OUT_F, W3, b3);
  float* out0 = (float*)d_out;
  float* out1 = out0 + (size_t)N * OUT_F;
  k_sigmoid_out<<<blocks((size_t)N * OUT_F, T), T, 0, stream>>>(agg, out0, out1, (size_t)N * OUT_F);
}
